// TripleScatterModule_84318797955303
// MI455X (gfx1250) — matrix operand order repaired and checked by reading, not compiled, not run on hardware
//
#include <hip/hip_runtime.h>

// Problem constants (match reference)
#define F_IN   16
#define HID    64
#define F_OUT  16
#define R_DIM  128
#define C_DIM  8192
#define M_DIM  8192
#define K1     48      // 3*F_IN
#define K1P    64      // padded K for GEMM1 (two 32-wide k-steps)
#define JPB    8       // j columns per block
#define THREADS 256
#define NWAVES  8

typedef __attribute__((ext_vector_type(16))) __bf16 v16bf;
typedef __attribute__((ext_vector_type(8)))  float  v8f;

// ---- WMMA fragment index helpers (CDNA5 ISA 7.12.2 layouts, wave32) ----
// 16-bit A matrix 16x32: lane l holds row m=l&15; slot s (s=2*vgpr+half):
//   k = (vgpr&3)*2 + half + (vgpr>=4 ? 16 : 0) + (l>=16 ? 8 : 0)
__device__ __forceinline__ int a_kofs(int s, int lane) {
    int v = s >> 1, h = s & 1;
    return ((v & 3) * 2 + h) + ((v >= 4) ? 16 : 0) + ((lane >= 16) ? 8 : 0);
}
// 16-bit B matrix 32x16: lane l holds col n=l&15; slot s: k = (s&7) + (s>=8 ? 16 : 0) + 8*(l>=16)
__device__ __forceinline__ int b_kofs(int s, int lane) {
    return (s & 7) + ((s >= 8) ? 16 : 0) + ((lane >= 16) ? 8 : 0);
}

__global__ __launch_bounds__(THREADS) void triple_scatter_mlp_kernel(
    const float* __restrict__ x,        // [F_IN, R, C]
    const float* __restrict__ w1,       // [HID, K1]
    const float* __restrict__ b1,       // [HID]
    const float* __restrict__ w2,       // [F_OUT, HID]
    const float* __restrict__ b2,       // [F_OUT]
    const int*   __restrict__ ind0,     // [L, 2]
    const int*   __restrict__ ind1,     // [L, 2]
    const int*   __restrict__ ind2,     // [L, 2]
    const int*   __restrict__ mix,      // [3, M]
    float*       __restrict__ out)      // [F_OUT, R, C], zero-initialized
{
    __shared__ __bf16 sW1[K1P][HID];            // [k][n]  8 KB
    __shared__ __bf16 sW2[HID][F_OUT];          // [k][n]  2 KB
    __shared__ __bf16 sH[NWAVES][16][HID + 8];  // per-wave relu buffer ~18 KB

    const int tid  = threadIdx.x;
    const int lane = tid & 31;
    const int wave = tid >> 5;
    const int bn   = lane & 15;            // n (col) for B/C/D fragments
    const int am   = lane & 15;            // m (row) for A fragments
    const int hi   = (lane >= 16) ? 8 : 0; // C/D row offset for upper half-wave

    // ---- Stage weights into LDS as bf16 (transposed to [k][n], K padded) ----
    for (int i = tid; i < K1P * HID; i += THREADS) {
        int k = i / HID, n = i % HID;
        float v = (k < K1) ? w1[n * K1 + k] : 0.0f;
        sW1[k][n] = (__bf16)v;
    }
    for (int i = tid; i < HID * F_OUT; i += THREADS) {
        int k = i / F_OUT, n = i % F_OUT;
        sW2[k][n] = (__bf16)w2[n * HID + k];
    }
    __syncthreads();

    // ---- Hoist B fragments + bias fragments into registers (loop-invariant) ----
    v16bf B1[2][4];   // [kstep][ntile] for GEMM1 (64x64 weights)
#pragma unroll
    for (int kk = 0; kk < 2; ++kk)
#pragma unroll
        for (int nt = 0; nt < 4; ++nt)
#pragma unroll
            for (int s = 0; s < 16; ++s)
                B1[kk][nt][s] = sW1[kk * 32 + b_kofs(s, lane)][nt * 16 + bn];

    v16bf B2[2];      // [kstep] for GEMM2 (64x16 weights)
#pragma unroll
    for (int kk = 0; kk < 2; ++kk)
#pragma unroll
        for (int s = 0; s < 16; ++s)
            B2[kk][s] = sW2[kk * 32 + b_kofs(s, lane)][bn];

    v8f bias1[4], bias2;
#pragma unroll
    for (int nt = 0; nt < 4; ++nt) {
        float bv = b1[nt * 16 + bn];
#pragma unroll
        for (int v = 0; v < 8; ++v) bias1[nt][v] = bv;
    }
    {
        float bv = b2[bn];
#pragma unroll
        for (int v = 0; v < 8; ++v) bias2[v] = bv;
    }

    const int r0 = wave * 16;     // this wave's row-tile base
    const int r  = r0 + am;       // this lane's row for A loads

    for (int jj = 0; jj < JPB; ++jj) {
        const int j  = blockIdx.x * JPB + jj;
        // effective gathered source columns (uniform across block)
        const int c0 = ind0[mix[0 * M_DIM + j] * 2 + 1];
        const int c1 = ind1[mix[1 * M_DIM + j] * 2 + 1];
        const int c2 = ind2[mix[2 * M_DIM + j] * 2 + 1];
        const int cv[3] = {c0, c1, c2};

        // ---- Build A fragments: concat(x[:,:,c0], x[:,:,c1], x[:,:,c2]) ----
        v16bf A[2];
#pragma unroll
        for (int f = 0; f < 2; ++f)
#pragma unroll
            for (int s = 0; s < 16; ++s) {
                int kg = f * 32 + a_kofs(s, lane);   // 0..63 (48..63 = zero pad)
                float val = 0.0f;
                if (kg < K1) {
                    int view = kg >> 4, feat = kg & 15;
                    val = x[(size_t)feat * (R_DIM * C_DIM) + (size_t)r * C_DIM + cv[view]];
                }
                A[f][s] = (__bf16)val;
            }

        // ---- GEMM1: [16,64] x [64,64] -> 4 accumulator tiles ----
        v8f acc[4];
#pragma unroll
        for (int nt = 0; nt < 4; ++nt) {
            acc[nt] = bias1[nt];
            acc[nt] = __builtin_amdgcn_wmma_f32_16x16x32_bf16(
                false, A[0], false, B1[0][nt], (short)0, acc[nt], false, false);
            acc[nt] = __builtin_amdgcn_wmma_f32_16x16x32_bf16(
                false, A[1], false, B1[1][nt], (short)0, acc[nt], false, false);
        }

        // ---- relu -> bf16 -> per-wave LDS (re-layout C/D -> A fragment) ----
        // fmaxf lowers to a single v_max_num_f32 (IEEE maximumNumber)
#pragma unroll
        for (int nt = 0; nt < 4; ++nt)
#pragma unroll
            for (int v = 0; v < 8; ++v) {
                float h = fmaxf(acc[nt][v], 0.0f);
                sH[wave][v + hi][nt * 16 + bn] = (__bf16)h;
            }
        // wave-internal LDS RAW: DS ops are in-order per wave; fence + drain
        asm volatile("s_wait_dscnt 0" ::: "memory");

        v16bf A2[2];
#pragma unroll
        for (int kk = 0; kk < 2; ++kk)
#pragma unroll
            for (int s = 0; s < 16; ++s)
                A2[kk][s] = sH[wave][am][kk * 32 + a_kofs(s, lane)];

        // ---- GEMM2: [16,64] x [64,16] -> [16,16] ----
        v8f d = bias2;
        d = __builtin_amdgcn_wmma_f32_16x16x32_bf16(
            false, A2[0], false, B2[0], (short)0, d, false, false);
        d = __builtin_amdgcn_wmma_f32_16x16x32_bf16(
            false, A2[1], false, B2[1], (short)0, d, false, false);

        // ---- Scatter-max into out[F_OUT, R, C] at columns c0,c1,c2 ----
        // out is zero-initialized and result = max(0, contributions), so a
        // signed-int atomic max on raw float bits is exact (negative floats
        // have negative int bit patterns < 0 == init).
#pragma unroll
        for (int v = 0; v < 8; ++v) {
            const int rr = r0 + v + hi;               // row
            const int o  = bn;                        // output feature
            const int bits = __float_as_int(d[v]);
            int* base = (int*)out + (size_t)o * (R_DIM * C_DIM) + (size_t)rr * C_DIM;
            atomicMax(base + c0, bits);
            if (c1 != c0)              atomicMax(base + c1, bits);
            if (c2 != c0 && c2 != c1)  atomicMax(base + c2, bits);
        }
    }
}

extern "C" void kernel_launch(void* const* d_in, const int* in_sizes, int n_in,
                              void* d_out, int out_size, void* d_ws, size_t ws_size,
                              hipStream_t stream) {
    const float* x    = (const float*)d_in[0];
    const float* w1   = (const float*)d_in[1];
    const float* b1   = (const float*)d_in[2];
    const float* w2   = (const float*)d_in[3];
    const float* b2   = (const float*)d_in[4];
    const int*   ind0 = (const int*)d_in[5];
    const int*   ind1 = (const int*)d_in[6];
    const int*   ind2 = (const int*)d_in[7];
    const int*   mix  = (const int*)d_in[8];
    float* out = (float*)d_out;

    // Scatter-max identity element: +0.0f (bit pattern 0) — capture-safe memset
    hipMemsetAsync(out, 0, (size_t)out_size * sizeof(float), stream);

    triple_scatter_mlp_kernel<<<M_DIM / JPB, THREADS, 0, stream>>>(
        x, w1, b1, w2, b2, ind0, ind1, ind2, mix, out);
}
